// MambaResidualBlock_40810779247401
// MI455X (gfx1250) — matrix operand order repaired and checked by reading, not compiled, not run on hardware
//
#include <hip/hip_runtime.h>
#include <cmath>

typedef __bf16 bf16_t;
typedef __attribute__((ext_vector_type(16))) __bf16 v16bf;
typedef __attribute__((ext_vector_type(8)))  __bf16 v8bf;
typedef __attribute__((ext_vector_type(8)))  float  v8f;
typedef __attribute__((ext_vector_type(4)))  int    v4i;

#define B_SZ 8
#define L_SZ 1024
#define D_SZ 512
#define DIN  1024
#define NST  16
#define DTR  32

#if __has_builtin(__builtin_amdgcn_global_load_async_to_lds_b128) && \
    __has_builtin(__builtin_amdgcn_s_wait_asynccnt)
#define USE_ASYNC_LDS 1
#else
#define USE_ASYNC_LDS 0
#endif

__device__ __forceinline__ float silu_f(float x) {
    return x * __builtin_amdgcn_rcpf(1.0f + __expf(-x));   // v_rcp_f32, no IEEE div
}
__device__ __forceinline__ float softplus_f(float x) {
    return (x > 20.0f) ? x : __logf(1.0f + __expf(x));     // v_log_f32 path
}

// 16-byte global->LDS tile copy; async-to-LDS when the toolchain exposes it.
// Builtin signature (from compiler diagnostic): (v4i AS1*, v4i AS3*, i32, i32).
__device__ __forceinline__ void copy16_g2l(const bf16_t* g, bf16_t* l) {
#if USE_ASYNC_LDS
    __builtin_amdgcn_global_load_async_to_lds_b128(
        (__attribute__((address_space(1))) v4i*)g,
        (__attribute__((address_space(3))) v4i*)l, 0, 0);
#else
    *(v8bf*)l = *(const v8bf*)g;
#endif
}
__device__ __forceinline__ void async_drain() {
#if USE_ASYNC_LDS
    __builtin_amdgcn_s_wait_asynccnt(0);
#endif
}

// ---------------------------------------------------------------------------
// Weight convert + transpose: fp32 [K][N] -> bf16 [N][K] (done once, tiny).
// ---------------------------------------------------------------------------
__global__ void wcvt_kernel(const float* __restrict__ in, bf16_t* __restrict__ out,
                            int K, int N) {
    int idx = blockIdx.x * 256 + threadIdx.x;
    if (idx >= K * N) return;
    int k = idx / N, n = idx % N;
    out[(size_t)n * K + k] = (bf16_t)in[idx];
}

// ---------------------------------------------------------------------------
// RMSNorm: one wave per row of 512; writes bf16 directly for GEMM1's A.
// ---------------------------------------------------------------------------
__global__ void rmsnorm_kernel(const float* __restrict__ x,
                               const float* __restrict__ w,
                               bf16_t* __restrict__ xn) {
    int wid  = threadIdx.x >> 5;
    int lane = threadIdx.x & 31;
    int row  = blockIdx.x * 8 + wid;
    const float* xr = x + (size_t)row * D_SZ;
    float v[16];
    float s = 0.f;
#pragma unroll
    for (int i = 0; i < 16; ++i) { v[i] = xr[lane + i * 32]; s += v[i] * v[i]; }
#pragma unroll
    for (int off = 16; off; off >>= 1) s += __shfl_xor(s, off, 32);
    float r = rsqrtf(s * (1.0f / D_SZ) + 1e-5f);
    bf16_t* xo = xn + (size_t)row * D_SZ;
#pragma unroll
    for (int i = 0; i < 16; ++i) xo[lane + i * 32] = (bf16_t)(v[i] * r * w[lane + i * 32]);
}

// ---------------------------------------------------------------------------
// bf16 WMMA GEMM, C = A(MxK,row) * B^T(NxK,row)^T. Macro tile 128x64 per
// 256-thread block; 8 waves in 4x2; each wave computes 32x32 = 4 WMMAs/K-step.
// Double-buffered LDS staging: async-to-LDS copies for tile i+1 are issued
// before computing tile i and only drained after the WMMAs, hiding global
// latency behind matrix math. 30 KB LDS of the 320 KB/WGP budget.
// Fragments: A lane m=l&15,half=l>>4: two b128 at [m][half*8], [m][16+half*8].
//            B lane n=l&15: two b128 at Bt [n][half*8], [n][16+half*8].
// MODE 0: silu split -> u_f32|u_bf16|silu(res)   MODE 1: f32 + bf16 copies
// MODE 2: softplus(v + bias)                     MODE 3: v + residual
// ---------------------------------------------------------------------------
template <int MODE>
__global__ void gemm_wmma_kernel(const bf16_t* __restrict__ A, int lda,
                                 const bf16_t* __restrict__ Bt, int ldbt,
                                 int K,
                                 float* __restrict__ O0,
                                 float* __restrict__ O1,
                                 bf16_t* __restrict__ O2,
                                 const float* __restrict__ EX, int ldo) {
    __shared__ __align__(16) bf16_t As[2][128][40];
    __shared__ __align__(16) bf16_t Bs[2][64][40];

    const int tid  = threadIdx.x;
    const int row0 = blockIdx.y * 128;
    const int col0 = blockIdx.x * 64;

    const int wid  = tid >> 5;
    const int lane = tid & 31;
    const int l15  = lane & 15;
    const int half = lane >> 4;
    const int mw   = (wid >> 1) * 32;   // {0,32,64,96}
    const int nw   = (wid & 1) * 32;    // {0,32}

    // A staging: 512 x 16B segments (128 rows x 4), 2 per thread.
    const int ar = tid >> 1;                       // row of first segment pair
    const int ao = (tid & 1) * 16;                 // 2 segs -> cols [ao, ao+16)
    // B staging: 256 x 16B segments (64 rows x 4), 1 per thread.
    const int br = tid >> 2, bo = (tid & 3) * 8;

    v8f acc[2][2] = {};

    const int NT = K >> 5;                         // K-tiles of 32

    // Prologue: stage tile 0 into buffer 0.
    copy16_g2l(A + (size_t)(row0 + ar) * lda + ao,     &As[0][ar][ao]);
    copy16_g2l(A + (size_t)(row0 + ar) * lda + ao + 8, &As[0][ar][ao + 8]);
    copy16_g2l(Bt + (size_t)(col0 + br) * ldbt + bo,   &Bs[0][br][bo]);
    async_drain();
    __syncthreads();

    for (int it = 0; it < NT; ++it) {
        const int cur = it & 1;
        // Issue next tile's copies into the idle buffer (overlaps with WMMAs).
        if (it + 1 < NT) {
            const int k1 = (it + 1) << 5;
            copy16_g2l(A + (size_t)(row0 + ar) * lda + k1 + ao,     &As[cur ^ 1][ar][ao]);
            copy16_g2l(A + (size_t)(row0 + ar) * lda + k1 + ao + 8, &As[cur ^ 1][ar][ao + 8]);
            copy16_g2l(Bt + (size_t)(col0 + br) * ldbt + k1 + bo,   &Bs[cur ^ 1][br][bo]);
        }

        union { v16bf v; v8bf h[2]; } ua[2], ub[2];
#pragma unroll
        for (int mi = 0; mi < 2; ++mi) {
            ua[mi].h[0] = *(const v8bf*)&As[cur][mw + mi * 16 + l15][half * 8];
            ua[mi].h[1] = *(const v8bf*)&As[cur][mw + mi * 16 + l15][16 + half * 8];
        }
#pragma unroll
        for (int ni = 0; ni < 2; ++ni) {
            ub[ni].h[0] = *(const v8bf*)&Bs[cur][nw + ni * 16 + l15][half * 8];
            ub[ni].h[1] = *(const v8bf*)&Bs[cur][nw + ni * 16 + l15][16 + half * 8];
        }
#pragma unroll
        for (int mi = 0; mi < 2; ++mi)
#pragma unroll
            for (int ni = 0; ni < 2; ++ni)
                acc[mi][ni] = __builtin_amdgcn_wmma_f32_16x16x32_bf16(
                    false, ua[mi].v, false, ub[ni].v, (short)0, acc[mi][ni], false, false);

        // Drain the in-flight next-tile copies only after compute, then rotate.
        async_drain();
        __syncthreads();
    }

    // Epilogue: c[r] = C[m = r + half*8][n = l15] within each 16x16 tile.
#pragma unroll
    for (int mi = 0; mi < 2; ++mi)
#pragma unroll
        for (int ni = 0; ni < 2; ++ni)
#pragma unroll
            for (int r = 0; r < 8; ++r) {
                const int m = row0 + mw + mi * 16 + half * 8 + r;
                const int n = col0 + nw + ni * 16 + l15;
                float v = acc[mi][ni][r];
                if (MODE == 0) {
                    float sv = silu_f(v);
                    if (n < DIN) {
                        O0[(size_t)m * DIN + n] = sv;          // u (fp32 for scan)
                        O2[(size_t)m * DIN + n] = (bf16_t)sv;  // u (bf16 for GEMM2)
                    } else {
                        O1[(size_t)m * DIN + (n - DIN)] = sv;  // silu(res)
                    }
                } else if (MODE == 1) {
                    O0[(size_t)m * ldo + n] = v;               // x_dbl fp32 (B,C)
                    O2[(size_t)m * ldo + n] = (bf16_t)v;       // x_dbl bf16 (dt GEMM)
                } else if (MODE == 2) {
                    O0[(size_t)m * ldo + n] = softplus_f(v + EX[n]);
                } else {
                    O0[(size_t)m * ldo + n] = v + EX[(size_t)m * ldo + n];
                }
            }
}

// ---------------------------------------------------------------------------
// Selective scan + fused gating. One lane per (batch, channel); 16 states in
// registers; B_t/C_t/reset broadcast via LDS. Writes gated output directly
// in bf16 (A operand of out_proj GEMM) and the final hidden state.
// ---------------------------------------------------------------------------
__global__ void scan_kernel(const float* __restrict__ u,
                            const float* __restrict__ delta,
                            const float* __restrict__ xdbl,
                            const float* __restrict__ rs,
                            const float* __restrict__ A_log,
                            const float* __restrict__ hidden_in,
                            const float* __restrict__ resg,
                            const float* __restrict__ Dp,
                            bf16_t* __restrict__ ysbf,
                            float* __restrict__ h_out) {
    const int b = blockIdx.x >> 2;
    const int d = ((blockIdx.x & 3) << 8) + threadIdx.x;

    float A[NST], h[NST];
#pragma unroll
    for (int n = 0; n < NST; ++n) {
        A[n] = -__expf(A_log[d * NST + n]);
        h[n] = hidden_in[(size_t)b * (DIN * NST) + d * NST + n];
    }
    const float Dv = Dp[d];

    __shared__ float sB[NST], sC[NST], sS;
    const float* xd   = xdbl + (size_t)b * L_SZ * 64;
    const size_t base = (size_t)b * L_SZ * DIN + d;

    for (int t = 0; t < L_SZ; ++t) {
        if (threadIdx.x < 16)       sB[threadIdx.x]      = xd[t * 64 + DTR + threadIdx.x];
        else if (threadIdx.x < 32)  sC[threadIdx.x - 16] = xd[t * 64 + DTR + NST + (threadIdx.x - 16)];
        else if (threadIdx.x == 32) sS = rs[b * L_SZ + t];
        __syncthreads();

        const float om = 1.0f - sS;
        const float dt = delta[base + (size_t)t * DIN];
        const float ut = u[base + (size_t)t * DIN];
        float y = 0.f;
#pragma unroll
        for (int n = 0; n < NST; ++n) {
            float dA = __expf(dt * A[n]);
            h[n] = dA * (h[n] * om) + dt * sB[n] * ut;
            y += h[n] * sC[n];
        }
        const float rv = resg[base + (size_t)t * DIN];
        ysbf[base + (size_t)t * DIN] = (bf16_t)((y + ut * Dv) * rv);
        __syncthreads();
    }
#pragma unroll
    for (int n = 0; n < NST; ++n)
        h_out[(size_t)b * (DIN * NST) + d * NST + n] = h[n];
}

// ---------------------------------------------------------------------------
extern "C" void kernel_launch(void* const* d_in, const int* in_sizes, int n_in,
                              void* d_out, int out_size, void* d_ws, size_t ws_size,
                              hipStream_t stream) {
    const float* x         = (const float*)d_in[0];
    const float* hidden    = (const float*)d_in[1];
    const float* rnn_start = (const float*)d_in[2];
    const float* norm_w    = (const float*)d_in[3];
    const float* in_proj_w = (const float*)d_in[4];
    const float* x_proj_w  = (const float*)d_in[5];
    const float* dt_proj_w = (const float*)d_in[6];
    const float* dt_proj_b = (const float*)d_in[7];
    const float* A_log     = (const float*)d_in[8];
    const float* D_param   = (const float*)d_in[9];
    const float* out_pw    = (const float*)d_in[10];

    float* out = (float*)d_out;                         // (8,1024,512)
    float* hT  = out + (size_t)B_SZ * L_SZ * D_SZ;      // (8,16384)

    const size_t MF = 1024ull * 1024ull;
    float* W      = (float*)d_ws;
    float* u_f32  = W;                  // 8M floats
    float* resg   = W + 8 * MF;         // 8M floats (silu(res))
    float* delta  = W + 16 * MF;        // 8M floats
    float* xdbl   = W + 24 * MF;        // 0.5M floats (8192x64)

    bf16_t* W2     = (bf16_t*)(W + 25 * MF);
    bf16_t* xn_bf  = W2;                      // 4M  (8192x512)
    bf16_t* u_bf   = W2 + 4  * MF;            // 8M  (8192x1024)
    bf16_t* xd_bf  = W2 + 12 * MF;            // 0.5M (8192x64)
    bf16_t* ys_bf  = W2 + 13 * MF;            // 8M  (8192x1024)
    bf16_t* inT    = W2 + 21 * MF;            // 2048x512  = 1M
    bf16_t* xpT    = inT + 1 * MF;            // 64x1024   = 64K
    bf16_t* dtT    = xpT + 64 * 1024;         // 1024x32   = 32K
    bf16_t* opT    = dtT + 32 * 1024;         // 512x1024  = 512K

    // 0. one-time weight convert+transpose (bf16, [N][K])
    wcvt_kernel<<<dim3((512 * 2048 + 255) / 256), dim3(256), 0, stream>>>(in_proj_w, inT, 512, 2048);
    wcvt_kernel<<<dim3((1024 * 64 + 255) / 256), dim3(256), 0, stream>>>(x_proj_w, xpT, 1024, 64);
    wcvt_kernel<<<dim3((32 * 1024 + 255) / 256), dim3(256), 0, stream>>>(dt_proj_w, dtT, 32, 1024);
    wcvt_kernel<<<dim3((1024 * 512 + 255) / 256), dim3(256), 0, stream>>>(out_pw, opT, 1024, 512);

    // 1. RMSNorm -> bf16
    rmsnorm_kernel<<<dim3(1024), dim3(256), 0, stream>>>(x, norm_w, xn_bf);

    // 2. in_proj: xn(8192x512) x W(512x2048) -> silu(u) fp32+bf16 | silu(res)
    gemm_wmma_kernel<0><<<dim3(2048 / 64, 8192 / 128), dim3(256), 0, stream>>>(
        xn_bf, D_SZ, inT, D_SZ, D_SZ, u_f32, resg, u_bf, nullptr, 0);

    // 3. x_proj: u(8192x1024) x W(1024x64) -> x_dbl fp32 + bf16
    gemm_wmma_kernel<1><<<dim3(64 / 64, 8192 / 128), dim3(256), 0, stream>>>(
        u_bf, DIN, xpT, DIN, DIN, xdbl, nullptr, xd_bf, nullptr, 64);

    // 4. dt: x_dbl[:, :32](8192x32) x W(32x1024) + b -> softplus -> delta
    gemm_wmma_kernel<2><<<dim3(1024 / 64, 8192 / 128), dim3(256), 0, stream>>>(
        xd_bf, 64, dtT, DTR, DTR, delta, nullptr, nullptr, dt_proj_b, DIN);

    // 5. scan + fused gating -> ys_bf, hT
    scan_kernel<<<dim3(32), dim3(256), 0, stream>>>(
        u_f32, delta, xdbl, rnn_start, A_log, hidden, resg, D_param, ys_bf, hT);

    // 6. out_proj + residual: ys(8192x1024) x W(1024x512) + x -> out
    gemm_wmma_kernel<3><<<dim3(512 / 64, 8192 / 128), dim3(256), 0, stream>>>(
        ys_bf, DIN, opT, DIN, DIN, out, nullptr, nullptr, x, D_SZ);
}
